// RoutingLayer_33311766348133
// MI455X (gfx1250) — matrix operand order repaired and checked by reading, not compiled, not run on hardware
//
#include <hip/hip_runtime.h>
#include <hip/hip_bf16.h>

// ---------------- problem constants ----------------
#define BB 64
#define HH 56
#define WW 56
#define CC 256
#define ROUTES 4
#define COUT 64
#define HP 27              // (56-3)/2+1
#define KFLAT (9 * CC)     // 2304
#define NCHUNK 8
#define ROWS_PER_CHUNK (HH / NCHUNK)  // 7
#define X_ELEMS ((size_t)BB * HH * WW * 64)  // gathered output elements

typedef __attribute__((ext_vector_type(16))) _Float16 v16h;
typedef __attribute__((ext_vector_type(8)))  float    v8f;
typedef __attribute__((ext_vector_type(4)))  float    v4f;

// ---------------------------------------------------------------------------
// Phase 1: 9-bin strided spatial reduction of the input.
// S[b, kh, kw, c] = sum over 27x27 grid of in[b, 2h'+kh, 2w'+kw, c].
// Separable: per row h, build rw[kw] (w-reduction), then fold into acc[kh][kw].
// Each input element is read from HBM exactly once, as b128 (one channel quad
// per lane -> 512 B per wave32 load instruction, clause-friendly for 23.3 TB/s).
// Grid: (NCHUNK, B), block: 64 threads (thread q owns channels 4q..4q+3).
// ---------------------------------------------------------------------------
__global__ void k_reduce9(const float* __restrict__ in, float* __restrict__ s_part) {
    const int q  = threadIdx.x;       // channel quad 0..63
    const int ch = blockIdx.x;        // row chunk 0..7
    const int b  = blockIdx.y;        // image 0..63

    v4f acc[3][3] = {};               // 9 x float4 accumulators

    const float* base = in + ((size_t)b * HH * WW) * CC + (size_t)q * 4;
    const int h0 = ch * ROWS_PER_CHUNK;

    for (int hr = 0; hr < ROWS_PER_CHUNK; ++hr) {
        const int h = h0 + hr;
        const float* row = base + (size_t)h * WW * CC;
        // prefetch next row while streaming this one (gfx1250 global_prefetch_b8)
        if (hr + 1 < ROWS_PER_CHUNK) __builtin_prefetch(row + (size_t)WW * CC, 0, 1);

        v4f rw0 = {}, rw1 = {}, rw2 = {};
#pragma unroll
        for (int w = 0; w < WW; w += 2) {
            const v4f ve = *(const v4f*)(row + (size_t)w * CC);        // even col
            const v4f vo = *(const v4f*)(row + (size_t)(w + 1) * CC);  // odd col
            if (w <= 52) rw0 += ve;   // kw = 0 : w' = w/2 <= 26
            if (w >= 2)  rw2 += ve;   // kw = 2 : w' = (w-2)/2 >= 0
            if (w <= 52) rw1 += vo;   // kw = 1 : w' = (w-1)/2 <= 26  (w+1 <= 53)
        }

        if ((h & 1) == 0) {
            if (h <= 52) { acc[0][0] += rw0; acc[0][1] += rw1; acc[0][2] += rw2; }
            if (h >= 2)  { acc[2][0] += rw0; acc[2][1] += rw1; acc[2][2] += rw2; }
        } else {
            if (h <= 53) { acc[1][0] += rw0; acc[1][1] += rw1; acc[1][2] += rw2; }
        }
    }

    float* dst = s_part + ((size_t)(b * NCHUNK + ch) * 9) * CC + (size_t)q * 4;
#pragma unroll
    for (int kh = 0; kh < 3; ++kh)
#pragma unroll
        for (int kw = 0; kw < 3; ++kw)
            *(v4f*)(dst + (size_t)(kh * 3 + kw) * CC) = acc[kh][kw];
}

// ---------------------------------------------------------------------------
// Phase 1b: fold chunk partials -> S[b][2304], float4 per thread,
// deterministic order.
// ---------------------------------------------------------------------------
__global__ void k_sum_chunks(const float* __restrict__ s_part, float* __restrict__ S) {
    const int i = blockIdx.x * blockDim.x + threadIdx.x;   // quad index
    if (i >= BB * KFLAT / 4) return;
    const int b = i / (KFLAT / 4);
    v4f s = {};
#pragma unroll
    for (int chk = 0; chk < NCHUNK; ++chk)
        s += *(const v4f*)(s_part + ((size_t)b * NCHUNK + chk) * KFLAT +
                           ((size_t)i % (KFLAT / 4)) * 4);
    *(v4f*)(S + (size_t)i * 4) = s;
}

// ---------------------------------------------------------------------------
// Phase 2: pooled(64x64) = (1/729) * S(64x2304) @ Wflat(2304x64) + conv_b
// via v_wmma_f32_16x16x32_f16. One block, 16 waves; wave (mi,ni) owns the
// 16x16 output tile (mi*16.., ni*16..); 72 k-steps of K=32.
// A layout (16-bit 16x32): lane L holds row M=L%16; elements t=0..7 -> K=t,
// t=8..15 -> K=8+t; hi lane half adds +8.   B layout (32x16): lane L holds
// col N=L%16; same K mapping as A.
// ---------------------------------------------------------------------------
__global__ void k_pool_gemm(const float* __restrict__ S,
                            const float* __restrict__ convw,   // 2304 x 64 row-major
                            const float* __restrict__ convb,
                            float* __restrict__ pooled) {
    const int tid  = threadIdx.x;
    const int wv   = tid >> 5;
    const int lane = tid & 31;
    const int mi   = wv >> 2;         // 0..3  (M tile)
    const int ni   = wv & 3;          // 0..3  (N tile)
    const int m    = lane & 15;
    const int hi   = lane >> 4;       // 0 = lanes 0-15, 1 = lanes 16-31

    v8f acc = {};
    const float* Srow = S + (size_t)(mi * 16 + m) * KFLAT;

    for (int kk = 0; kk < 72; ++kk) {
        const int kbase = kk * 32;
        v16h av, bv;
#pragma unroll
        for (int t = 0; t < 16; ++t) {
            const int Ka = kbase + ((t < 8) ? t : (8 + t)) + hi * 8;
            av[t] = (_Float16)Srow[Ka];
            const int Kb = kbase + ((t < 8) ? t : (8 + t)) + hi * 8;
            bv[t] = (_Float16)convw[(size_t)Kb * COUT + ni * 16 + m];
        }
        acc = __builtin_amdgcn_wmma_f32_16x16x32_f16(
            /*neg_a=*/false, av, /*neg_b=*/false, bv,
            /*c_mod=*/(short)0, acc, /*reuse_a=*/false, /*reuse_b=*/false);
    }

    const float inv = 1.0f / (float)(HP * HP);   // 1/729
    const int n = ni * 16 + m;
    const float bias = convb[n];
#pragma unroll
    for (int j = 0; j < 8; ++j) {
        const int row = mi * 16 + j + hi * 8;    // C/D layout: VGPR j, lane half
        pooled[row * COUT + n] = acc[j] * inv + bias;
    }
}

// ---------------------------------------------------------------------------
// Phase 3: logits = pooled @ fc_w + fc_b ; route = argmax (first max wins).
// One thread per image.
// ---------------------------------------------------------------------------
__global__ void k_logits_route(const float* __restrict__ pooled,
                               const float* __restrict__ fcw,   // 64 x 4 row-major
                               const float* __restrict__ fcb,
                               float* __restrict__ logits_out,
                               int* __restrict__ route) {
    const int b = threadIdx.x;
    if (b >= BB) return;
    float l0 = fcb[0], l1 = fcb[1], l2 = fcb[2], l3 = fcb[3];
    const float* p = pooled + (size_t)b * COUT;
#pragma unroll 8
    for (int o = 0; o < COUT; ++o) {
        const float v = p[o];
        l0 += v * fcw[o * 4 + 0];
        l1 += v * fcw[o * 4 + 1];
        l2 += v * fcw[o * 4 + 2];
        l3 += v * fcw[o * 4 + 3];
    }
    logits_out[b * 4 + 0] = l0;
    logits_out[b * 4 + 1] = l1;
    logits_out[b * 4 + 2] = l2;
    logits_out[b * 4 + 3] = l3;
    int r = 0; float best = l0;
    if (l1 > best) { best = l1; r = 1; }
    if (l2 > best) { best = l2; r = 2; }
    if (l3 > best) { best = l3; r = 3; }
    route[b] = r;
}

// ---------------------------------------------------------------------------
// Phase 4: channel-group gather, float4 vectorized (16B per lane, coalesced).
// out[b,h,w,j] = in[b,h,w, route[b]*64 + j], j in [0,64).
// Grid: (196, B), 256 threads; 196*256 float4 = 50176 = 56*56*64/4 per image.
// ---------------------------------------------------------------------------
__global__ void k_gather(const float* __restrict__ in,
                         const int* __restrict__ route,
                         float* __restrict__ out) {
    const int b = blockIdx.y;
    const int cbase = route[b] * 64;                    // uniform -> scalar load
    const int idx = blockIdx.x * blockDim.x + threadIdx.x;  // 0..50175
    const int p  = idx >> 4;                            // pixel within image
    const int j4 = idx & 15;                            // float4 slot in group
    const float4* src = (const float4*)(in + ((size_t)(b * (HH * WW) + p) * CC + cbase));
    float4* dst = (float4*)(out + (size_t)(b * (HH * WW) + p) * 64);
    dst[j4] = src[j4];
}

// ---------------------------------------------------------------------------
extern "C" void kernel_launch(void* const* d_in, const int* in_sizes, int n_in,
                              void* d_out, int out_size, void* d_ws, size_t ws_size,
                              hipStream_t stream) {
    const float* in    = (const float*)d_in[0];   // (64,56,56,256)
    const float* convw = (const float*)d_in[1];   // (3,3,256,64) == (2304,64)
    const float* convb = (const float*)d_in[2];   // (64,)
    const float* fcw   = (const float*)d_in[3];   // (64,4)
    const float* fcb   = (const float*)d_in[4];   // (4,)
    float* out = (float*)d_out;

    // workspace layout
    char* ws = (char*)d_ws;
    float* s_part = (float*)ws;                                      // 64*8*2304 f32 (4.7 MB)
    float* S      = (float*)(ws + (size_t)BB * NCHUNK * KFLAT * 4);  // 64*2304
    float* pooled = (float*)(ws + (size_t)(BB * NCHUNK * KFLAT + BB * KFLAT) * 4);
    int*   route  = (int*)  (ws + (size_t)(BB * NCHUNK * KFLAT + BB * KFLAT + BB * COUT) * 4);

    float* logits = out + X_ELEMS;  // x first, then logits (tuple order)

    k_reduce9<<<dim3(NCHUNK, BB), 64, 0, stream>>>(in, s_part);
    k_sum_chunks<<<(BB * KFLAT / 4 + 255) / 256, 256, 0, stream>>>(s_part, S);
    k_pool_gemm<<<1, 512, 0, stream>>>(S, convw, convb, pooled);
    k_logits_route<<<1, 64, 0, stream>>>(pooled, fcw, fcb, logits, route);
    k_gather<<<dim3((HH * WW * 64 / 4) / 256, BB), 256, 0, stream>>>(in, route, out);
}
